// MMDLoss_62405874811366
// MI455X (gfx1250) — matrix operand order repaired and checked by reading, not compiled, not run on hardware
//
#include <hip/hip_runtime.h>
#include <hip/hip_bf16.h>
#include <stdint.h>

// ---------------------------------------------------------------------------
// MMD multi-bandwidth loss, MI455X (gfx1250):
//   comb (8192x256) -> bf16 (4MB, L2-resident).
//   Gram matrix recomputed in two fused passes (histogram -> sigma -> exp sums)
//   instead of materializing the 256MB dsq matrix.
//   GEMM engine: 8-wave block covers 128x256 output; 384 source rows staged
//   into 96KB LDS (half-K per phase) via global_load_async_to_lds_b128;
//   each wave computes a 64x64 register tile = 16 x v_wmma_f32_16x16x32_bf16
//   per K-step, fragments read from LDS (ds_load_b128).
// ---------------------------------------------------------------------------

#define NROW  4096
#define DIM   256
#define M2    8192            // combined rows (x;y)
#define BI    128             // block i-rows
#define BJ    256             // block j-rows
#define KP    128             // K elements staged per phase
#define NBJ   (M2 / BJ)       // 32
#define NBLK  ((M2 / BI) * (M2 / BJ))   // 64*32 = 2048 blocks
#define NBINS 8192
#define BINW  0.25f           // histogram covers dsq in [0, 2048)
#define INV_BINW 4.0f

typedef __attribute__((ext_vector_type(16))) __bf16         v16bf;
typedef __attribute__((ext_vector_type(8)))  float          v8f;
typedef __attribute__((ext_vector_type(8)))  unsigned short v8us;

union FragBF { v16bf v; v8us u[2]; };

__device__ __forceinline__ unsigned short f2bf_rne(float f) {
    unsigned u = __float_as_uint(f);
    unsigned r = u + 0x7FFFu + ((u >> 16) & 1u);
    return (unsigned short)(r >> 16);
}

// ---- build bf16 combined matrix [x;y] ------------------------------------
__global__ void __launch_bounds__(256)
convert_bf16(const float* __restrict__ x, const float* __restrict__ y,
             unsigned short* __restrict__ abf) {
    int idx = blockIdx.x * 256 + threadIdx.x;      // 0 .. M2*DIM-1
    int row = idx >> 8;
    int col = idx & 255;
    float f = (row < NROW) ? x[row * DIM + col] : y[(row - NROW) * DIM + col];
    abf[idx] = f2bf_rne(f);
}

// ---- per-row squared norms (from the bf16 data, self-consistent) ---------
__global__ void __launch_bounds__(256)
row_sqnorm(const unsigned short* __restrict__ abf, float* __restrict__ sq) {
    __shared__ float red[256];
    int row = blockIdx.x;
    int t = threadIdx.x;
    float f = __uint_as_float(((unsigned)abf[(size_t)row * DIM + t]) << 16);
    red[t] = f * f;
    __syncthreads();
    for (int s = 128; s > 0; s >>= 1) {
        if (t < s) red[t] += red[t + s];
        __syncthreads();
    }
    if (t == 0) sq[row] = red[0];
}

// ---- stage 384 rows x KP columns into LDS via CDNA5 async loads ----------
// tileK layout: rows 0..BI-1 = i-rows, rows BI..BI+BJ-1 = j-rows; KP bf16/row.
__device__ __forceinline__ void stage_phase(const unsigned short* __restrict__ abf,
                                            unsigned short* tileK,
                                            unsigned ibase, unsigned jbase, int kp) {
    const int tid = threadIdx.x;
    // (BI+BJ)*KP*2 bytes / 16B chunks = 6144 chunks; 24 per thread
#pragma unroll
    for (int c = 0; c < 24; ++c) {
        int chunk = tid + c * 256;
        int row = chunk >> 4;                 // 0..383
        int ch  = chunk & 15;                 // 16-byte chunk within 256B row
        unsigned gr = (row < BI) ? (ibase + (unsigned)row)
                                 : (jbase + (unsigned)(row - BI));
        const unsigned short* g = abf + (size_t)gr * DIM + (size_t)kp * KP + ch * 8;
        // generic shared pointer: low 32 bits are the LDS byte offset
        unsigned lofs = (unsigned)(uintptr_t)(&tileK[row * KP + ch * 8]);
        asm volatile("global_load_async_to_lds_b128 %0, %1, off"
                     :: "v"(lofs), "v"(g) : "memory");
    }
    asm volatile("s_wait_asynccnt 0" ::: "memory");
}

// ---- 64x64 register-tile Gram block from LDS-staged rows ------------------
// wave (wi,wj) inside block; c[it][jt] accumulates 16x16 f32 tiles.
__device__ __forceinline__ void gram_block(const unsigned short* tileK,
                                           unsigned wi, unsigned wj,
                                           unsigned lane, v8f c[4][4]) {
    const unsigned n = lane & 15u;
    const unsigned h = lane >> 4;
#pragma unroll
    for (int ks = 0; ks < 4; ++ks) {
        const int kk0 = ks * 32;
        FragBF b[4];
#pragma unroll
        for (int jt = 0; jt < 4; ++jt) {
            const unsigned short* Br = &tileK[(BI + wj * 64 + jt * 16 + n) * KP];
            b[jt].u[0] = *(const v8us*)(Br + kk0 + 16 * h);
            b[jt].u[1] = *(const v8us*)(Br + kk0 + 16 * h + 8);
        }
#pragma unroll
        for (int it = 0; it < 4; ++it) {
            const unsigned short* Ar = &tileK[(wi * 64 + it * 16 + n) * KP];
            FragBF a;
            a.u[0] = *(const v8us*)(Ar + kk0 + 16 * h);
            a.u[1] = *(const v8us*)(Ar + kk0 + 16 * h + 8);
#pragma unroll
            for (int jt = 0; jt < 4; ++jt) {
                c[it][jt] = __builtin_amdgcn_wmma_f32_16x16x32_bf16(
                        false, a.v, false, b[jt].v, (short)0, c[it][jt],
                        false, false);
            }
        }
    }
}

// ---- pass 1: fused GEMM + dsq histogram over strict upper triangle -------
__global__ void __launch_bounds__(256)
mmd_hist(const unsigned short* __restrict__ abf, const float* __restrict__ sq,
         unsigned int* __restrict__ hist) {
    __shared__ __attribute__((aligned(16))) unsigned short tileK[(BI + BJ) * KP];
    __shared__ unsigned int lh[NBINS];
    for (int i = threadIdx.x; i < NBINS; i += 256) lh[i] = 0u;
    __syncthreads();

    const unsigned ibase = (blockIdx.x / NBJ) * BI;
    const unsigned jbase = (blockIdx.x % NBJ) * BJ;

    if (ibase < jbase + BJ) {            // block intersects strict upper triangle
        const unsigned wave = threadIdx.x >> 5;
        const unsigned wi = wave >> 2, wj = wave & 3u;
        const unsigned lane = threadIdx.x & 31u;
        const unsigned n = lane & 15u;
        const unsigned h = lane >> 4;

        v8f c[4][4];
#pragma unroll
        for (int it = 0; it < 4; ++it)
#pragma unroll
            for (int jt = 0; jt < 4; ++jt) c[it][jt] = (v8f){};

        for (int kp = 0; kp < 2; ++kp) {
            stage_phase(abf, tileK, ibase, jbase, kp);
            __syncthreads();
            gram_block(tileK, wi, wj, lane, c);
            __syncthreads();
        }

        float sqjv[4];
#pragma unroll
        for (int jt = 0; jt < 4; ++jt) sqjv[jt] = sq[jbase + wj * 64 + jt * 16 + n];

#pragma unroll
        for (int it = 0; it < 4; ++it) {
#pragma unroll
            for (int e = 0; e < 8; ++e) {
                unsigned i = ibase + wi * 64 + it * 16 + e + 8 * h;
                float sqi = sq[i];
#pragma unroll
                for (int jt = 0; jt < 4; ++jt) {
                    unsigned j = jbase + wj * 64 + jt * 16 + n;
                    if (i < j) {
                        float dsq = fmaxf(sqi + sqjv[jt] - 2.0f * c[it][jt][e], 0.0f);
                        int bin = (int)(dsq * INV_BINW);
                        bin = bin < NBINS ? bin : (NBINS - 1);
                        atomicAdd(&lh[bin], 1u);
                    }
                }
            }
        }
    }
    __syncthreads();
    for (int i = threadIdx.x; i < NBINS; i += 256)
        if (lh[i]) atomicAdd(&hist[i], lh[i]);
}

// ---- median of dsq -> sigma (median of sqrt == sqrt of median) -----------
__global__ void median_from_hist(const unsigned int* __restrict__ hist,
                                 float* __restrict__ sigma) {
    if (threadIdx.x == 0) {
        const unsigned long long total = (unsigned long long)M2 * (M2 - 1) / 2;
        const unsigned long long need  = total / 2;
        unsigned long long cum = 0;
        int b = NBINS - 1;
        float frac = 0.5f;
        for (int i = 0; i < NBINS; ++i) {
            unsigned cnt = hist[i];
            if (cum + cnt > need) {
                b = i;
                frac = (float)(need - cum) / (float)(cnt ? cnt : 1u);
                break;
            }
            cum += cnt;
        }
        sigma[0] = sqrtf(((float)b + frac) * BINW);
    }
}

// ---- pass 2: fused GEMM + three-bandwidth exp sums -----------------------
// sums layout: [xx0,xx1,xx2, yy0,yy1,yy2, xy0,xy1,xy2]
__global__ void __launch_bounds__(256)
mmd_kernel_sums(const unsigned short* __restrict__ abf,
                const float* __restrict__ sq,
                const float* __restrict__ sigma,
                float* __restrict__ sums) {
    __shared__ __attribute__((aligned(16))) unsigned short tileK[(BI + BJ) * KP];
    __shared__ float red[256];

    const float sg = sigma[0];
    const float bw0 = 0.5f * sg, bw1 = sg, bw2 = 2.0f * sg;
    const float inv0 = 1.0f / (2.0f * bw0 * bw0);
    const float inv1 = 1.0f / (2.0f * bw1 * bw1);
    const float inv2 = 1.0f / (2.0f * bw2 * bw2);

    const unsigned ibase = (blockIdx.x / NBJ) * BI;
    const unsigned jbase = (blockIdx.x % NBJ) * BJ;
    const unsigned wave = threadIdx.x >> 5;
    const unsigned wi = wave >> 2, wj = wave & 3u;
    const unsigned lane = threadIdx.x & 31u;
    const unsigned n = lane & 15u;
    const unsigned h = lane >> 4;

    v8f c[4][4];
#pragma unroll
    for (int it = 0; it < 4; ++it)
#pragma unroll
        for (int jt = 0; jt < 4; ++jt) c[it][jt] = (v8f){};

    for (int kp = 0; kp < 2; ++kp) {
        stage_phase(abf, tileK, ibase, jbase, kp);
        __syncthreads();
        gram_block(tileK, wi, wj, lane, c);
        __syncthreads();
    }

    float acc[9];
#pragma unroll
    for (int t = 0; t < 9; ++t) acc[t] = 0.0f;

    float sqjv[4];
    bool  jxv[4];
#pragma unroll
    for (int jt = 0; jt < 4; ++jt) {
        unsigned j = jbase + wj * 64 + jt * 16 + n;
        sqjv[jt] = sq[j];
        jxv[jt]  = j < NROW;
    }

#pragma unroll
    for (int it = 0; it < 4; ++it) {
#pragma unroll
        for (int e = 0; e < 8; ++e) {
            unsigned i = ibase + wi * 64 + it * 16 + e + 8 * h;
            float sqi = sq[i];
            bool ix = i < NROW;
#pragma unroll
            for (int jt = 0; jt < 4; ++jt) {
                float dsq = fmaxf(sqi + sqjv[jt] - 2.0f * c[it][jt][e], 0.0f);
                int base = ix ? (jxv[jt] ? 0 : 6)     // xx -> 0, xy -> 6
                              : (jxv[jt] ? 9 : 3);    // yx -> skip, yy -> 3
                if (base < 9) {
                    acc[base + 0] += __expf(-dsq * inv0);
                    acc[base + 1] += __expf(-dsq * inv1);
                    acc[base + 2] += __expf(-dsq * inv2);
                }
            }
        }
    }

    // block-wide reduction, one atomic per accumulator per block
#pragma unroll
    for (int t = 0; t < 9; ++t) {
        red[threadIdx.x] = acc[t];
        __syncthreads();
        for (int s = 128; s > 0; s >>= 1) {
            if (threadIdx.x < s) red[threadIdx.x] += red[threadIdx.x + s];
            __syncthreads();
        }
        if (threadIdx.x == 0 && red[0] != 0.0f) atomicAdd(&sums[t], red[0]);
        __syncthreads();
    }
}

// ---- combine sums -> scalar loss -----------------------------------------
__global__ void finalize(const float* __restrict__ sums, float* __restrict__ out) {
    if (threadIdx.x == 0 && blockIdx.x == 0) {
        const float nn1 = (float)NROW * (float)(NROW - 1);
        const float nm  = (float)NROW * (float)NROW;
        float loss = 0.0f;
        for (int p = 0; p < 2; ++p) {
            const float* s = sums + p * 9;
            for (int k = 0; k < 3; ++k) {
                float exx = (s[0 + k] - (float)NROW) / nn1;  // subtract trace
                float eyy = (s[3 + k] - (float)NROW) / nn1;
                float exy = s[6 + k] / nm;
                loss += exx + eyy - 2.0f * exy;
            }
        }
        out[0] = loss;
    }
}

// ---------------------------------------------------------------------------
extern "C" void kernel_launch(void* const* d_in, const int* in_sizes, int n_in,
                              void* d_out, int out_size, void* d_ws, size_t ws_size,
                              hipStream_t stream) {
    const float* xA = (const float*)d_in[0];
    const float* yA = (const float*)d_in[1];
    const float* xC = (const float*)d_in[2];
    const float* yC = (const float*)d_in[3];

    char* ws = (char*)d_ws;
    size_t off = 0;
    unsigned short* abf0 = (unsigned short*)(ws + off); off += (size_t)M2 * DIM * 2; // 4 MB
    unsigned short* abf1 = (unsigned short*)(ws + off); off += (size_t)M2 * DIM * 2; // 4 MB
    float*    sq0   = (float*)(ws + off);    off += (size_t)M2 * 4;
    float*    sq1   = (float*)(ws + off);    off += (size_t)M2 * 4;
    unsigned* hist0 = (unsigned*)(ws + off); off += (size_t)NBINS * 4;
    unsigned* hist1 = (unsigned*)(ws + off); off += (size_t)NBINS * 4;
    float*    sig0  = (float*)(ws + off);    off += 256;
    float*    sig1  = (float*)(ws + off);    off += 256;
    float*    sums  = (float*)(ws + off);    off += 18 * 4;   // 2 pairs x 9

    hipMemsetAsync(hist0, 0, (size_t)NBINS * 4, stream);
    hipMemsetAsync(hist1, 0, (size_t)NBINS * 4, stream);
    hipMemsetAsync(sums,  0, 18 * 4, stream);

    dim3 blk(256);
    convert_bf16<<<(M2 * DIM) / 256, blk, 0, stream>>>(xA, yA, abf0);
    convert_bf16<<<(M2 * DIM) / 256, blk, 0, stream>>>(xC, yC, abf1);
    row_sqnorm<<<M2, blk, 0, stream>>>(abf0, sq0);
    row_sqnorm<<<M2, blk, 0, stream>>>(abf1, sq1);

    mmd_hist<<<NBLK, blk, 0, stream>>>(abf0, sq0, hist0);
    mmd_hist<<<NBLK, blk, 0, stream>>>(abf1, sq1, hist1);
    median_from_hist<<<1, 32, 0, stream>>>(hist0, sig0);
    median_from_hist<<<1, 32, 0, stream>>>(hist1, sig1);
    mmd_kernel_sums<<<NBLK, blk, 0, stream>>>(abf0, sq0, sig0, sums);
    mmd_kernel_sums<<<NBLK, blk, 0, stream>>>(abf1, sq1, sig1, sums + 9);
    finalize<<<1, 32, 0, stream>>>(sums, (float*)d_out);
}
